// MLA_31903017074827
// MI455X (gfx1250) — matrix operand order repaired and checked by reading, not compiled, not run on hardware
//
#include <hip/hip_runtime.h>
#include <math.h>

typedef __attribute__((ext_vector_type(16))) _Float16 v16h;
typedef __attribute__((ext_vector_type(8)))  float    v8f;

#define BB 8
#define LL 512
#define DM 1024
#define NH 16
#define HD 64
#define RK 128
#define MM (BB*LL)   // 4096

// ---------------------------------------------------------------------------
// Fragment loaders (CDNA5 wave32 WMMA layouts, 16-bit data, 16x16x32)
// A 16x32: lanes 0-15 -> row=lane, K={0..7,16..23}; lanes 16-31 -> row=lane-16,
//          K={8..15,24..31}.  p = row base (+k0); halfw = lane>>4.
// B 32x16: lanes 0-15 -> col=lane, K={0..7,16..23}; lanes 16-31 -> col=lane-16, K={8..15,24..31}.
//          For C = A*W^T with W stored (N,K) row-major, "column n" of B is W row n.
// ---------------------------------------------------------------------------
__device__ inline v16h load_afrag(const _Float16* __restrict__ p, int halfw) {
    union { v16h v; _Float16 e[16]; } u;
    const _Float16* q = p + halfw * 8;
#pragma unroll
    for (int i = 0; i < 8; i++) { u.e[i] = q[i]; u.e[8 + i] = q[16 + i]; }
    return u.v;
}
__device__ inline v16h load_bfrag(const _Float16* __restrict__ p, int halfw) {
    union { v16h v; _Float16 e[16]; } u;
    const _Float16* q = p + halfw * 8;
#pragma unroll
    for (int i = 0; i < 8; i++) { u.e[i] = q[i]; u.e[8 + i] = q[16 + i]; }
    return u.v;
}

// ---------------------------------------------------------------------------
// f32 -> f16 conversion
// ---------------------------------------------------------------------------
__global__ void cvt_f32_f16(const float* __restrict__ src, _Float16* __restrict__ dst, int n) {
    int i = blockIdx.x * 256 + threadIdx.x;
    if (i < n) dst[i] = (_Float16)src[i];
}

// ---------------------------------------------------------------------------
// Wide GEMM: C(MxN) = A(MxK,f16) * W(NxK,f16)^T + bias(f32)
// Block = 256 threads = 8 waves, arranged 4(M) x 2(N).
// Wave computes 32(M) x 64(N) = 8 WMMA tiles -> 8 wmmas per 6 fragment loads.
// Block tile: 128(M) x 128(N).  Requires M%128==0, N%128==0, K%32==0.
// ---------------------------------------------------------------------------
__global__ __launch_bounds__(256) void gemm_bias_f16_wide(
    const _Float16* __restrict__ A, const _Float16* __restrict__ Bw,
    const float* __restrict__ bias, _Float16* __restrict__ Oh,
    float* __restrict__ Of, int M, int N, int K)
{
    const int w     = threadIdx.x >> 5;
    const int lane  = threadIdx.x & 31;
    const int halfw = lane >> 4;
    const int l16   = lane & 15;
    const int wm    = w & 3;    // 4 M positions
    const int wn    = w >> 2;   // 2 N positions
    const int tileM = blockIdx.y * 128 + wm * 32;
    const int tileN = blockIdx.x * 128 + wn * 64;

    const _Float16* arow0 = A  + (size_t)(tileM + l16) * K;
    const _Float16* arow1 = A  + (size_t)(tileM + 16 + l16) * K;
    const _Float16* brow0 = Bw + (size_t)(tileN + l16) * K;
    const _Float16* brow1 = Bw + (size_t)(tileN + 16 + l16) * K;
    const _Float16* brow2 = Bw + (size_t)(tileN + 32 + l16) * K;
    const _Float16* brow3 = Bw + (size_t)(tileN + 48 + l16) * K;

    v8f acc[2][4];
#pragma unroll
    for (int mi = 0; mi < 2; mi++)
#pragma unroll
        for (int ni = 0; ni < 4; ni++)
#pragma unroll
            for (int r = 0; r < 8; r++) acc[mi][ni][r] = 0.f;

    for (int k0 = 0; k0 < K; k0 += 32) {
        v16h a0 = load_afrag(arow0 + k0, halfw);
        v16h a1 = load_afrag(arow1 + k0, halfw);
        v16h b0 = load_bfrag(brow0 + k0, halfw);
        v16h b1 = load_bfrag(brow1 + k0, halfw);
        v16h b2 = load_bfrag(brow2 + k0, halfw);
        v16h b3 = load_bfrag(brow3 + k0, halfw);
        acc[0][0] = __builtin_amdgcn_wmma_f32_16x16x32_f16(false, a0, false, b0, (short)0, acc[0][0], false, false);
        acc[0][1] = __builtin_amdgcn_wmma_f32_16x16x32_f16(false, a0, false, b1, (short)0, acc[0][1], false, false);
        acc[0][2] = __builtin_amdgcn_wmma_f32_16x16x32_f16(false, a0, false, b2, (short)0, acc[0][2], false, false);
        acc[0][3] = __builtin_amdgcn_wmma_f32_16x16x32_f16(false, a0, false, b3, (short)0, acc[0][3], false, false);
        acc[1][0] = __builtin_amdgcn_wmma_f32_16x16x32_f16(false, a1, false, b0, (short)0, acc[1][0], false, false);
        acc[1][1] = __builtin_amdgcn_wmma_f32_16x16x32_f16(false, a1, false, b1, (short)0, acc[1][1], false, false);
        acc[1][2] = __builtin_amdgcn_wmma_f32_16x16x32_f16(false, a1, false, b2, (short)0, acc[1][2], false, false);
        acc[1][3] = __builtin_amdgcn_wmma_f32_16x16x32_f16(false, a1, false, b3, (short)0, acc[1][3], false, false);
    }

#pragma unroll
    for (int mi = 0; mi < 2; mi++) {
#pragma unroll
        for (int ni = 0; ni < 4; ni++) {
            const int col = tileN + ni * 16 + l16;
            const float bv = bias[col];
#pragma unroll
            for (int r = 0; r < 8; r++) {
                const int row = tileM + mi * 16 + halfw * 8 + r;
                float v = acc[mi][ni][r] + bv;
                size_t idx = (size_t)row * N + col;
                if (Of) Of[idx] = v; else Oh[idx] = (_Float16)v;
            }
        }
    }
}

// ---------------------------------------------------------------------------
// Narrow GEMM (kept for the rank-128 latent projection: preserves occupancy
// when N is small).  Wave computes 16(M) x 32(N); block tile 32 x 128.
// ---------------------------------------------------------------------------
__global__ __launch_bounds__(256) void gemm_bias_f16_small(
    const _Float16* __restrict__ A, const _Float16* __restrict__ Bw,
    const float* __restrict__ bias, _Float16* __restrict__ Oh,
    int M, int N, int K)
{
    const int w     = threadIdx.x >> 5;
    const int lane  = threadIdx.x & 31;
    const int halfw = lane >> 4;
    const int l16   = lane & 15;
    const int tileM = blockIdx.y * 32 + (w & 1) * 16;
    const int tileN = blockIdx.x * 128 + (w >> 1) * 32;

    const _Float16* arow  = A  + (size_t)(tileM + l16) * K;
    const _Float16* brow0 = Bw + (size_t)(tileN + l16) * K;
    const _Float16* brow1 = Bw + (size_t)(tileN + 16 + l16) * K;

    v8f acc0, acc1;
#pragma unroll
    for (int i = 0; i < 8; i++) { acc0[i] = 0.f; acc1[i] = 0.f; }

    for (int k0 = 0; k0 < K; k0 += 32) {
        v16h a  = load_afrag(arow + k0, halfw);
        v16h b0 = load_bfrag(brow0 + k0, halfw);
        v16h b1 = load_bfrag(brow1 + k0, halfw);
        acc0 = __builtin_amdgcn_wmma_f32_16x16x32_f16(false, a, false, b0, (short)0, acc0, false, false);
        acc1 = __builtin_amdgcn_wmma_f32_16x16x32_f16(false, a, false, b1, (short)0, acc1, false, false);
    }

    const int col0 = tileN + l16, col1 = tileN + 16 + l16;
    const float bv0 = bias[col0], bv1 = bias[col1];
#pragma unroll
    for (int r = 0; r < 8; r++) {
        const int row = tileM + halfw * 8 + r;
        Oh[(size_t)row * N + col0] = (_Float16)(acc0[r] + bv0);
        Oh[(size_t)row * N + col1] = (_Float16)(acc1[r] + bv1);
    }
}

// ---------------------------------------------------------------------------
// RoPE on Q (full head dim rotated, ROPE_DIM == HEAD_DIM == 64) + transpose:
// in  Qh  (B, L, H, 64)  f16   ->  out Qr (B, H, L, 64) f16
// ---------------------------------------------------------------------------
__global__ void rope_kernel(const _Float16* __restrict__ Qh, _Float16* __restrict__ Qr) {
    int idx = blockIdx.x * 256 + threadIdx.x;
    if (idx >= BB * LL * NH * (HD / 2)) return;
    int i = idx & 31;          // pair index 0..31
    int h = (idx >> 5) & 15;
    int l = (idx >> 9) & 511;
    int b = idx >> 18;
    const _Float16* src = Qh + (((size_t)(b * LL + l) * NH + h) * HD);
    float xr = (float)src[2 * i];
    float xi = (float)src[2 * i + 1];
    float theta = __powf(10000.0f, -(float)i / 32.0f);
    float ang = (float)l * theta;
    float c = cosf(ang), s = sinf(ang);
    _Float16* dst = Qr + ((((size_t)b * NH + h) * LL + l) * HD);
    dst[2 * i]     = (_Float16)(xr * c - xi * s);
    dst[2 * i + 1] = (_Float16)(xr * s + xi * c);
}

// ---------------------------------------------------------------------------
// Split KV (4096 x 2048 f16) into:
//   Kh (B,H,L,64)  (keys x dims)
//   Vt (B,H,64,L)  (dims x keys, transposed so P*V B-fragments are contiguous)
// ---------------------------------------------------------------------------
__global__ void splitkv_kernel(const _Float16* __restrict__ KV,
                               _Float16* __restrict__ Kh, _Float16* __restrict__ Vt) {
    int idx = blockIdx.x * 256 + threadIdx.x;
    if (idx >= MM * 2 * DM) return;
    int n = idx & 2047;
    int m = idx >> 11;
    int l = m & 511;
    int b = m >> 9;
    _Float16 val = KV[idx];
    if (n < DM) {
        int h = n >> 6, d = n & 63;
        Kh[(((size_t)(b * NH + h) * LL + l) * HD) + d] = val;
    } else {
        int n2 = n - DM;
        int h = n2 >> 6, d = n2 & 63;
        Vt[(((size_t)(b * NH + h) * HD + d) * LL) + l] = val;
    }
}

// ---------------------------------------------------------------------------
// Flash attention: block = (b,h, 128-query tile), 8 waves x 16 rows each.
// Online softmax, 32 keys per iteration:
//   S  = Q(16x64) * K^T          -> 4 wmma
//   ctx += P(16x32) * V(32x64)   -> 4 wmma
// P is bounced through LDS to convert C-fragment layout -> A-fragment layout.
// ---------------------------------------------------------------------------
__global__ __launch_bounds__(256) void attn_flash(
    const _Float16* __restrict__ Qr, const _Float16* __restrict__ Kh,
    const _Float16* __restrict__ Vt, const unsigned char* __restrict__ mask,
    _Float16* __restrict__ Cxh)
{
    __shared__ _Float16 Plds[8][16][32];   // 8 KB, one 16x32 P tile per wave

    const int bh = blockIdx.x;
    const int b = bh >> 4, h = bh & 15;
    const int w = threadIdx.x >> 5, lane = threadIdx.x & 31;
    const int halfw = lane >> 4, l16 = lane & 15;
    const int q0 = blockIdx.y * 128 + w * 16;

    const _Float16* Qb = Qr + (size_t)bh * LL * HD;
    const _Float16* Kb = Kh + (size_t)bh * LL * HD;
    const _Float16* Vb = Vt + (size_t)bh * HD * LL;
    const unsigned char* mrow = mask + b * LL;

    const _Float16* qrow = Qb + (size_t)(q0 + l16) * HD;
    v16h qa0 = load_afrag(qrow, halfw);
    v16h qa1 = load_afrag(qrow + 32, halfw);

    v8f o0, o1, o2, o3;
    float mr[8], lr[8];
#pragma unroll
    for (int r = 0; r < 8; r++) {
        o0[r] = 0.f; o1[r] = 0.f; o2[r] = 0.f; o3[r] = 0.f;
        mr[r] = -1e30f; lr[r] = 0.f;
    }

    for (int n0 = 0; n0 < LL; n0 += 32) {
        v8f s0, s1;
#pragma unroll
        for (int r = 0; r < 8; r++) { s0[r] = 0.f; s1[r] = 0.f; }
        const _Float16* kr0 = Kb + (size_t)(n0 + l16) * HD;
        const _Float16* kr1 = Kb + (size_t)(n0 + 16 + l16) * HD;
        {
            v16h kb00 = load_bfrag(kr0, halfw);
            v16h kb01 = load_bfrag(kr0 + 32, halfw);
            s0 = __builtin_amdgcn_wmma_f32_16x16x32_f16(false, qa0, false, kb00, (short)0, s0, false, false);
            s0 = __builtin_amdgcn_wmma_f32_16x16x32_f16(false, qa1, false, kb01, (short)0, s0, false, false);
            v16h kb10 = load_bfrag(kr1, halfw);
            v16h kb11 = load_bfrag(kr1 + 32, halfw);
            s1 = __builtin_amdgcn_wmma_f32_16x16x32_f16(false, qa0, false, kb10, (short)0, s1, false, false);
            s1 = __builtin_amdgcn_wmma_f32_16x16x32_f16(false, qa1, false, kb11, (short)0, s1, false, false);
        }

        const bool msk0 = mrow[n0 + l16] != 0;
        const bool msk1 = mrow[n0 + 16 + l16] != 0;

        // online softmax per row (row = halfw*8 + r; col = lane&15)
#pragma unroll
        for (int r = 0; r < 8; r++) {
            float f0 = msk0 ? s0[r] * 0.125f : -1e30f;   // 1/sqrt(64)
            float f1 = msk1 ? s1[r] * 0.125f : -1e30f;
            float t = fmaxf(f0, f1);
            t = fmaxf(t, __shfl_xor(t, 1, 32));
            t = fmaxf(t, __shfl_xor(t, 2, 32));
            t = fmaxf(t, __shfl_xor(t, 4, 32));
            t = fmaxf(t, __shfl_xor(t, 8, 32));
            float nm = fmaxf(mr[r], t);
            float alpha = __expf(mr[r] - nm);
            float p0 = __expf(f0 - nm);
            float p1 = __expf(f1 - nm);
            float rs = p0 + p1;
            rs += __shfl_xor(rs, 1, 32);
            rs += __shfl_xor(rs, 2, 32);
            rs += __shfl_xor(rs, 4, 32);
            rs += __shfl_xor(rs, 8, 32);
            mr[r] = nm;
            lr[r] = lr[r] * alpha + rs;
            o0[r] *= alpha; o1[r] *= alpha; o2[r] *= alpha; o3[r] *= alpha;
            Plds[w][halfw * 8 + r][l16]      = (_Float16)p0;
            Plds[w][halfw * 8 + r][16 + l16] = (_Float16)p1;
        }
        __syncthreads();

        // re-load P as A fragment (16x32)
        v16h pa;
        {
            union { v16h v; _Float16 e[16]; } u;
#pragma unroll
            for (int i = 0; i < 8; i++) {
                u.e[i]     = Plds[w][l16][halfw * 8 + i];
                u.e[8 + i] = Plds[w][l16][16 + halfw * 8 + i];
            }
            pa = u.v;
        }
        __syncthreads();

        // ctx += P * V ; V^T stored (dim, key) so B-frags are contiguous
#pragma unroll
        for (int c = 0; c < 4; c++) {
            const _Float16* vp = Vb + (size_t)(c * 16 + l16) * LL + n0 + halfw * 8;
            union { v16h v; _Float16 e[16]; } u;
#pragma unroll
            for (int i = 0; i < 16; i++) u.e[i] = vp[(i < 8) ? i : i + 8];
            v8f* op = (c == 0) ? &o0 : (c == 1) ? &o1 : (c == 2) ? &o2 : &o3;
            *op = __builtin_amdgcn_wmma_f32_16x16x32_f16(false, pa, false, u.v, (short)0, *op, false, false);
        }
    }

    // normalize and write ctx as (B, L, H*64) f16
#pragma unroll
    for (int r = 0; r < 8; r++) {
        const int q = q0 + halfw * 8 + r;
        const float inv = 1.0f / lr[r];
        size_t base = ((size_t)(b * LL + q) * DM) + h * HD;
        Cxh[base + 0 * 16 + l16] = (_Float16)(o0[r] * inv);
        Cxh[base + 1 * 16 + l16] = (_Float16)(o1[r] * inv);
        Cxh[base + 2 * 16 + l16] = (_Float16)(o2[r] * inv);
        Cxh[base + 3 * 16 + l16] = (_Float16)(o3[r] * inv);
    }
}

// ---------------------------------------------------------------------------
// Host launcher
// ---------------------------------------------------------------------------
extern "C" void kernel_launch(void* const* d_in, const int* in_sizes, int n_in,
                              void* d_out, int out_size, void* d_ws, size_t ws_size,
                              hipStream_t stream) {
    (void)in_sizes; (void)n_in; (void)out_size; (void)ws_size;
    const float* x       = (const float*)d_in[0];
    const unsigned char* mask = (const unsigned char*)d_in[1];
    const float* Wkva_w  = (const float*)d_in[2];
    const float* Wkva_b  = (const float*)d_in[3];
    const float* Wkvb_w  = (const float*)d_in[4];
    const float* Wkvb_b  = (const float*)d_in[5];
    const float* Wq_w    = (const float*)d_in[6];
    const float* Wq_b    = (const float*)d_in[7];
    const float* Wo_w    = (const float*)d_in[8];
    const float* Wo_b    = (const float*)d_in[9];
    float* out = (float*)d_out;

    // workspace carve-up
    char* ws = (char*)d_ws;
    size_t off = 0;
    _Float16* Xh    = (_Float16*)(ws + off); off += (size_t)MM * DM * 2;        // 8 MB
    _Float16* Wqh   = (_Float16*)(ws + off); off += (size_t)DM * DM * 2;        // 2 MB
    _Float16* Wkvah = (_Float16*)(ws + off); off += (size_t)RK * DM * 2;        // 256 KB
    _Float16* Wkvbh = (_Float16*)(ws + off); off += (size_t)2 * DM * RK * 2;    // 512 KB
    _Float16* Woh   = (_Float16*)(ws + off); off += (size_t)DM * DM * 2;        // 2 MB
    _Float16* Qh    = (_Float16*)(ws + off); off += (size_t)MM * DM * 2;        // 8 MB
    _Float16* Qrope = (_Float16*)(ws + off); off += (size_t)MM * DM * 2;        // 8 MB
    _Float16* Lath  = (_Float16*)(ws + off); off += (size_t)MM * RK * 2;        // 1 MB
    _Float16* KVh   = (_Float16*)(ws + off); off += (size_t)MM * 2 * DM * 2;    // 16 MB
    _Float16* Kh    = (_Float16*)(ws + off); off += (size_t)MM * DM * 2;        // 8 MB
    _Float16* Vt    = (_Float16*)(ws + off); off += (size_t)MM * DM * 2;        // 8 MB
    _Float16* Cxh   = (_Float16*)(ws + off); off += (size_t)MM * DM * 2;        // 8 MB

    // 1) convert inputs / weights to f16
    cvt_f32_f16<<<(MM * DM + 255) / 256, 256, 0, stream>>>(x, Xh, MM * DM);
    cvt_f32_f16<<<(DM * DM + 255) / 256, 256, 0, stream>>>(Wq_w, Wqh, DM * DM);
    cvt_f32_f16<<<(RK * DM + 255) / 256, 256, 0, stream>>>(Wkva_w, Wkvah, RK * DM);
    cvt_f32_f16<<<(2 * DM * RK + 255) / 256, 256, 0, stream>>>(Wkvb_w, Wkvbh, 2 * DM * RK);
    cvt_f32_f16<<<(DM * DM + 255) / 256, 256, 0, stream>>>(Wo_w, Woh, DM * DM);

    // 2) Q = X Wq^T + b   (4096 x 1024, K=1024)
    gemm_bias_f16_wide<<<dim3(DM / 128, MM / 128), 256, 0, stream>>>(
        Xh, Wqh, Wq_b, Qh, nullptr, MM, DM, DM);

    // 3) RoPE + transpose to (B,H,L,64)
    rope_kernel<<<(BB * LL * NH * (HD / 2) + 255) / 256, 256, 0, stream>>>(Qh, Qrope);

    // 4) latent = X Wkva^T + b   (4096 x 128, K=1024) -- narrow kernel for occupancy
    gemm_bias_f16_small<<<dim3(RK / 128, MM / 32), 256, 0, stream>>>(
        Xh, Wkvah, Wkva_b, Lath, MM, RK, DM);

    // 5) KV = latent Wkvb^T + b   (4096 x 2048, K=128)
    gemm_bias_f16_wide<<<dim3(2 * DM / 128, MM / 128), 256, 0, stream>>>(
        Lath, Wkvbh, Wkvb_b, KVh, nullptr, MM, 2 * DM, RK);

    // 6) split K / V (V transposed per head)
    splitkv_kernel<<<(MM * 2 * DM + 255) / 256, 256, 0, stream>>>(KVh, Kh, Vt);

    // 7) flash attention -> ctx (B,L,1024) f16
    attn_flash<<<dim3(BB * NH, LL / 128), 256, 0, stream>>>(Qrope, Kh, Vt, mask, Cxh);

    // 8) out = ctx Wo^T + b  (f32 output)
    gemm_bias_f16_wide<<<dim3(DM / 128, MM / 128), 256, 0, stream>>>(
        Cxh, Woh, Wo_b, nullptr, out, MM, DM, DM);
}
